// FusedLlamaMLPForQuantizedModel_1726576854713
// MI455X (gfx1250) — matrix operand order repaired and checked by reading, not compiled, not run on hardware
//
#include <hip/hip_runtime.h>

// ---------------------------------------------------------------------------
// Fused GPTQ-int4 Llama MLP for gfx1250 (MI455X): bf16 WMMA GEMMs with
// inline 4-bit dequant into LDS. wave32; V_WMMA_F32_16X16X32_BF16.
// K staged 64-deep per barrier (2 WMMA sub-steps), fast-rcp SiLU epilogue,
// global_prefetch on packed weights, async global->LDS staging (ASYNCcnt)
// for the bf16 activation tile in the down projection.
// ---------------------------------------------------------------------------

#define HID   4096
#define INTER 11008

typedef __attribute__((ext_vector_type(16))) __bf16 v16bf;
typedef __attribute__((ext_vector_type(8)))  __bf16 v8bf;
typedef __attribute__((ext_vector_type(8)))  float  v8f;

__device__ __forceinline__ v16bf join16(v8bf lo, v8bf hi) {
    v16bf r;
#pragma unroll
    for (int i = 0; i < 8; ++i) { r[i] = lo[i]; r[i + 8] = hi[i]; }
    return r;
}

// Dequant 8 packed 4-bit values: (w - (z+1)) * s  ==  w*s - zs  (zs precomputed)
__device__ __forceinline__ void dequant8(unsigned q, float s, float zs, __bf16* dst) {
#pragma unroll
    for (int j = 0; j < 8; ++j) {
        float w = (float)((q >> (4 * j)) & 15u);
        dst[j] = (__bf16)(w * s - zs);
    }
}

// CDNA5 async global->LDS copy, 16B per lane (tracked by ASYNCcnt).
__device__ __forceinline__ void async_copy_b128(unsigned lds_addr, const void* gaddr) {
    asm volatile("global_load_async_to_lds_b128 %0, %1, off"
                 :: "v"(lds_addr), "v"((unsigned long long)(size_t)gaddr)
                 : "memory");
}
__device__ __forceinline__ void wait_asynccnt0() {
    asm volatile("s_wait_asynccnt 0x0" ::: "memory");
}

// ---------------------------------------------------------------------------
// Kernel 1: h = silu(x @ Wg) * (x @ Wu), h stored bf16 (M x INTER)
// block: 256 threads = 8 waves; tile 128(M) x 64(N); K staged 64-deep
// wave (id w): M-slot = (w&3)*32, N-slot = (w>>2)*32 -> 2x2 WMMA frags x2 mats
// ---------------------------------------------------------------------------
__global__ __launch_bounds__(256)
void gateup_silu_kernel(const float* __restrict__ x,
                        const int*   __restrict__ gqw, const float* __restrict__ gsc,
                        const int*   __restrict__ gqz,
                        const int*   __restrict__ uqw, const float* __restrict__ usc,
                        const int*   __restrict__ uqz,
                        __bf16*      __restrict__ hout)
{
    constexpr int BM = 128, BN = 64, BK = 64, KP = BK + 8;   // pad vs bank conflicts
    __shared__ __bf16 As [BM][KP];    // row-major   (m, k)
    __shared__ __bf16 Bgs[BN][KP];    // column-major(n, k)
    __shared__ __bf16 Bus[BN][KP];

    const int t    = threadIdx.x;
    const int lane = t & 31;
    const int wave = t >> 5;
    const int l16  = lane & 15;
    const int lh   = lane >> 4;       // 0/1 half-wave

    const int mBase = blockIdx.x * BM;
    const int nBase = blockIdx.y * BN;
    const int mw    = (wave & 3) * 32;
    const int nw    = (wave >> 2) * 32;

    const v8f vzero = {};
    v8f accG[2][2], accU[2][2];
#pragma unroll
    for (int i = 0; i < 2; ++i)
#pragma unroll
        for (int j = 0; j < 2; ++j) { accG[i][j] = vzero; accU[i][j] = vzero; }

    // A staging map: 2 threads per row, 32 f32 each
    const int arow = t >> 1;
    const int aoff = (t & 1) * 32;
    const float* __restrict__ xrow = x + (size_t)(mBase + arow) * HID;

    // B dequant map: thread -> (column in tile, two 8-deep K sub-blocks)
    const int bc  = t & 63;
    const int br  = t >> 6;                 // 0..3
    const int n   = nBase + bc;
    const int zsh = (n & 7) * 4;

    for (int kk = 0; kk < HID; kk += BK) {
        // ---- stage A (f32 -> bf16), 64 K deep ----
        const float4* xp = (const float4*)(xrow + kk + aoff);
        __bf16* ap = &As[arow][aoff];
#pragma unroll
        for (int c = 0; c < 8; ++c) {
            float4 f = xp[c];
            ap[4*c+0] = (__bf16)f.x; ap[4*c+1] = (__bf16)f.y;
            ap[4*c+2] = (__bf16)f.z; ap[4*c+3] = (__bf16)f.w;
        }

        // ---- inline GPTQ dequant of gate & up K-tiles (g_idx == k/128) ----
        const int g  = kk >> 7;               // 128|BK -> constant per stage
        const int kr = (kk >> 3) + br;        // first int32 row, second is +4
        {
            float    s  = gsc[(size_t)g * INTER + n];
            unsigned z  = (((unsigned)gqz[(size_t)g * (INTER / 8) + (n >> 3)]) >> zsh) & 15u;
            float    zs = (float)(z + 1u) * s;
            unsigned q0 = (unsigned)gqw[(size_t)kr * INTER + n];
            unsigned q1 = (unsigned)gqw[(size_t)(kr + 4) * INTER + n];
            dequant8(q0, s, zs, &Bgs[bc][br * 8]);
            dequant8(q1, s, zs, &Bgs[bc][32 + br * 8]);
        }
        {
            float    s  = usc[(size_t)g * INTER + n];
            unsigned z  = (((unsigned)uqz[(size_t)g * (INTER / 8) + (n >> 3)]) >> zsh) & 15u;
            float    zs = (float)(z + 1u) * s;
            unsigned q0 = (unsigned)uqw[(size_t)kr * INTER + n];
            unsigned q1 = (unsigned)uqw[(size_t)(kr + 4) * INTER + n];
            dequant8(q0, s, zs, &Bus[bc][br * 8]);
            dequant8(q1, s, zs, &Bus[bc][32 + br * 8]);
        }

        // prefetch next stage's packed-weight rows (global_prefetch_b8)
        if (kk + BK < HID) {
            __builtin_prefetch(&gqw[(size_t)(kr + 8) * INTER + n], 0, 1);
            __builtin_prefetch(&uqw[(size_t)(kr + 8) * INTER + n], 0, 1);
        }
        __syncthreads();

        // ---- two 32-deep WMMA sub-steps per LDS stage ----
#pragma unroll
        for (int ks = 0; ks < BK; ks += 32) {
            v16bf a[2], bg[2], bu[2];
#pragma unroll
            for (int mi = 0; mi < 2; ++mi) {
                const __bf16* p = &As[mw + mi * 16 + l16][ks];
                a[mi] = join16(*(const v8bf*)(p + 8 * lh), *(const v8bf*)(p + 16 + 8 * lh));
            }
#pragma unroll
            for (int ni = 0; ni < 2; ++ni) {
                const __bf16* pg = &Bgs[nw + ni * 16 + l16][ks];
                bg[ni] = join16(*(const v8bf*)(pg + 8 * lh), *(const v8bf*)(pg + 16 + 8 * lh));
                const __bf16* pu = &Bus[nw + ni * 16 + l16][ks];
                bu[ni] = join16(*(const v8bf*)(pu + 8 * lh), *(const v8bf*)(pu + 16 + 8 * lh));
            }
#pragma unroll
            for (int mi = 0; mi < 2; ++mi)
#pragma unroll
                for (int ni = 0; ni < 2; ++ni) {
                    accG[mi][ni] = __builtin_amdgcn_wmma_f32_16x16x32_bf16(
                        false, a[mi], false, bg[ni], (short)0, accG[mi][ni], false, false);
                    accU[mi][ni] = __builtin_amdgcn_wmma_f32_16x16x32_bf16(
                        false, a[mi], false, bu[ni], (short)0, accU[mi][ni], false, false);
                }
        }
        __syncthreads();
    }

    // ---- epilogue: h = silu(g) * u, fast rcp (C/D layout: VGPR v -> M = v+8*lh) ----
#pragma unroll
    for (int mi = 0; mi < 2; ++mi)
#pragma unroll
        for (int ni = 0; ni < 2; ++ni) {
            const int gm0 = mBase + mw + mi * 16 + 8 * lh;
            const int gn  = nBase + nw + ni * 16 + l16;
#pragma unroll
            for (int v = 0; v < 8; ++v) {
                float gv = accG[mi][ni][v];
                float uv = accU[mi][ni][v];
                float e  = __expf(-gv);
                float hv = gv * __builtin_amdgcn_rcpf(1.0f + e) * uv;
                hout[(size_t)(gm0 + v) * INTER + gn] = (__bf16)hv;
            }
        }
}

// ---------------------------------------------------------------------------
// Kernel 2: out = h @ Wd   (M x INTER) x (INTER x HID) -> f32
// A tile staged with GLOBAL_LOAD_ASYNC_TO_LDS_B128 (ASYNCcnt path)
// ---------------------------------------------------------------------------
__global__ __launch_bounds__(256)
void down_kernel(const __bf16* __restrict__ hin,
                 const int*    __restrict__ dqw, const float* __restrict__ dsc,
                 const int*    __restrict__ dqz,
                 float*        __restrict__ out)
{
    constexpr int BM = 128, BN = 64, BK = 64, KP = BK + 8;
    __shared__ __bf16 As[BM][KP];
    __shared__ __bf16 Bs[BN][KP];

    const int t    = threadIdx.x;
    const int lane = t & 31;
    const int wave = t >> 5;
    const int l16  = lane & 15;
    const int lh   = lane >> 4;

    const int mBase = blockIdx.x * BM;
    const int nBase = blockIdx.y * BN;
    const int mw    = (wave & 3) * 32;
    const int nw    = (wave >> 2) * 32;

    const v8f vzero = {};
    v8f acc[2][2];
#pragma unroll
    for (int i = 0; i < 2; ++i)
#pragma unroll
        for (int j = 0; j < 2; ++j) acc[i][j] = vzero;

    const int arow = t >> 1;
    const int aoff = (t & 1) * 32;
    const __bf16* __restrict__ hrow = hin + (size_t)(mBase + arow) * INTER;
    const unsigned aldsbase = (unsigned)(size_t)&As[arow][aoff];   // low 32b = LDS offset

    const int bc  = t & 63;
    const int br  = t >> 6;
    const int n   = nBase + bc;
    const int zsh = (n & 7) * 4;

    for (int kk = 0; kk < INTER; kk += BK) {
        // stage A (already bf16): 64B per thread via async global->LDS DMA
        const __bf16* gp = hrow + kk + aoff;
        async_copy_b128(aldsbase +  0, gp +  0);
        async_copy_b128(aldsbase + 16, gp +  8);
        async_copy_b128(aldsbase + 32, gp + 16);
        async_copy_b128(aldsbase + 48, gp + 24);

        const int g  = kk >> 7;
        const int kr = (kk >> 3) + br;
        {
            float    s  = dsc[(size_t)g * HID + n];
            unsigned z  = (((unsigned)dqz[(size_t)g * (HID / 8) + (n >> 3)]) >> zsh) & 15u;
            float    zs = (float)(z + 1u) * s;
            unsigned q0 = (unsigned)dqw[(size_t)kr * HID + n];
            unsigned q1 = (unsigned)dqw[(size_t)(kr + 4) * HID + n];
            dequant8(q0, s, zs, &Bs[bc][br * 8]);
            dequant8(q1, s, zs, &Bs[bc][32 + br * 8]);
        }
        if (kk + BK < INTER)
            __builtin_prefetch(&dqw[(size_t)(kr + 8) * HID + n], 0, 1);

        wait_asynccnt0();          // A-tile DMA complete (compiler can't track asm)
        __syncthreads();

#pragma unroll
        for (int ks = 0; ks < BK; ks += 32) {
            v16bf a[2], b[2];
#pragma unroll
            for (int mi = 0; mi < 2; ++mi) {
                const __bf16* p = &As[mw + mi * 16 + l16][ks];
                a[mi] = join16(*(const v8bf*)(p + 8 * lh), *(const v8bf*)(p + 16 + 8 * lh));
            }
#pragma unroll
            for (int ni = 0; ni < 2; ++ni) {
                const __bf16* p = &Bs[nw + ni * 16 + l16][ks];
                b[ni] = join16(*(const v8bf*)(p + 8 * lh), *(const v8bf*)(p + 16 + 8 * lh));
            }
#pragma unroll
            for (int mi = 0; mi < 2; ++mi)
#pragma unroll
                for (int ni = 0; ni < 2; ++ni)
                    acc[mi][ni] = __builtin_amdgcn_wmma_f32_16x16x32_bf16(
                        false, a[mi], false, b[ni], (short)0, acc[mi][ni], false, false);
        }
        __syncthreads();
    }

#pragma unroll
    for (int mi = 0; mi < 2; ++mi)
#pragma unroll
        for (int ni = 0; ni < 2; ++ni) {
            const int gm0 = mBase + mw + mi * 16 + 8 * lh;
            const int gn  = nBase + nw + ni * 16 + l16;
#pragma unroll
            for (int v = 0; v < 8; ++v)
                out[(size_t)(gm0 + v) * HID + gn] = acc[mi][ni][v];
        }
}

// ---------------------------------------------------------------------------
extern "C" void kernel_launch(void* const* d_in, const int* in_sizes, int n_in,
                              void* d_out, int out_size, void* d_ws, size_t ws_size,
                              hipStream_t stream) {
    const float* x   = (const float*)d_in[0];
    const int*   gqw = (const int*)  d_in[1];
    const float* gsc = (const float*)d_in[2];
    const int*   gqz = (const int*)  d_in[3];
    // d_in[4] = gate_g_idx (deterministic k/128 -> computed in-kernel)
    const int*   uqw = (const int*)  d_in[5];
    const float* usc = (const float*)d_in[6];
    const int*   uqz = (const int*)  d_in[7];
    // d_in[8] = up_g_idx
    const int*   dqw = (const int*)  d_in[9];
    const float* dsc = (const float*)d_in[10];
    const int*   dqz = (const int*)  d_in[11];
    // d_in[12] = down_g_idx

    float*  out = (float*) d_out;
    __bf16* h   = (__bf16*)d_ws;          // M x INTER bf16 intermediate

    const int M = in_sizes[0] / HID;      // 8192

    dim3 blk(256);
    dim3 g1(M / 128, INTER / 64);
    gateup_silu_kernel<<<g1, blk, 0, stream>>>(x, gqw, gsc, gqz, uqw, usc, uqz, h);

    dim3 g2(M / 128, HID / 64);
    down_kernel<<<g2, blk, 0, stream>>>(h, dqw, dsc, dqz, out);
}
